// ImageEncoder_40527311405226
// MI455X (gfx1250) — matrix operand order repaired and checked by reading, not compiled, not run on hardware
//
#include <hip/hip_runtime.h>
#include <math.h>

// ---------------------------------------------------------------------------
// Types for CDNA5 WMMA (gfx1250, wave32)
// ---------------------------------------------------------------------------
typedef __bf16 v16bf __attribute__((ext_vector_type(16)));
typedef float  v8f   __attribute__((ext_vector_type(8)));

#define ACT_NONE     0
#define ACT_SOFTPLUS 1

__device__ __forceinline__ float silu_f(float x) {
    return x / (1.f + __expf(-x));
}

// ---------------------------------------------------------------------------
// WMMA GEMM:  C[M,N] = act( A[M,K] * W[N,K]^T + bias[N] ) + resid[M,N]
//   Block: 256 threads = 8 waves.  Block tile: 128 (M) x NT*16 (N), K-step 32.
//   NT = 8 (wide, 2x operand reuse) or NT = 4 (skinny N).
//   W tile staged in LDS (bf16), double-buffered: next tile's global loads
//   overlap current tile's WMMAs; one barrier per K-step.  B-fragment loads
//   software-pipelined one tile ahead of the WMMA consuming them.
//   Fragment layouts used here (contiguous-K order, same for A and B):
//     A 16x32 bf16: lane<16 row M=lane, contiguous K kb..kb+15, kb=0;
//                   lanes>=16 same rows, kb=16.
//     B 32x16 bf16: lane holds col N=lane&15, contiguous K range 16*(lane>>4).
//     D f32:        elem r -> row 8*(lane>>4)+r, col lane&15.
// ---------------------------------------------------------------------------
template <int NT>
__global__ __launch_bounds__(256)
void gemm_wmma_kernel(const float* __restrict__ A, int lda,
                      const float* __restrict__ W,
                      const float* __restrict__ bias,
                      const float* __restrict__ resid, int ldr,
                      float* __restrict__ C, int ldc,
                      int M, int N, int K, int act)
{
    __shared__ __align__(32) __bf16 Bs[2][NT * 16][32];

    const int tid  = threadIdx.x;
    const int lane = tid & 31;
    const int wave = tid >> 5;
    const int mBase  = blockIdx.y * 128 + wave * 16;
    const int nBlock = blockIdx.x * (NT * 16);

    const int KT = (K + 31) >> 5;

    // cooperative staging: each thread moves 8-consecutive-k runs of W rows
    auto stage = [&](int kt, int buf) {
        const int k0 = kt << 5;
        const bool kFull = (k0 + 32 <= K);
#pragma unroll
        for (int e = tid; e < NT * 64; e += 256) {
            const int row = e >> 2;
            const int kc  = (e & 3) << 3;
            const int gn  = nBlock + row;
            __align__(16) __bf16 tmp[8];
            if ((gn < N) && kFull) {
                const float* __restrict__ src = W + (size_t)gn * K + k0 + kc;
#pragma unroll
                for (int j = 0; j < 8; ++j) tmp[j] = (__bf16)src[j];
            } else {
#pragma unroll
                for (int j = 0; j < 8; ++j) {
                    const int gk = k0 + kc + j;
                    tmp[j] = (gn < N && gk < K) ? (__bf16)W[(size_t)gn * K + gk]
                                                : (__bf16)0.f;
                }
            }
            *reinterpret_cast<uint4*>(&Bs[buf][row][kc]) =
                *reinterpret_cast<const uint4*>(tmp);
        }
    };

    v8f acc[NT];
#pragma unroll
    for (int t = 0; t < NT; ++t)
#pragma unroll
        for (int i = 0; i < 8; ++i) acc[t][i] = 0.f;

    const int mRow = mBase + (lane & 15);
    const int mm   = (mRow < M) ? mRow : (M - 1);
    const int kb0  = (lane < 16) ? 0 : 16;
    const float* __restrict__ Arow = A + (size_t)mm * lda;
    const int bn   = lane & 15;
    const int bkb  = (lane >> 4) << 4;   // 0 or 16

    stage(0, 0);
    __syncthreads();

    for (int kt = 0; kt < KT; ++kt) {
        const int buf = kt & 1;
        if (kt + 1 < KT) stage(kt + 1, buf ^ 1);   // overlap next tile

        const int k0 = kt << 5;
        v16bf afrag;
        if (k0 + 32 <= K) {
#pragma unroll
            for (int j = 0; j < 8; ++j) afrag[j]     = (__bf16)Arow[k0 + kb0 + j];
#pragma unroll
            for (int j = 0; j < 8; ++j) afrag[8 + j] = (__bf16)Arow[k0 + 8 + kb0 + j];
        } else {
#pragma unroll
            for (int j = 0; j < 8; ++j) {
                const int k = k0 + kb0 + j;
                afrag[j] = (k < K) ? (__bf16)Arow[k] : (__bf16)0.f;
            }
#pragma unroll
            for (int j = 0; j < 8; ++j) {
                const int k = k0 + 8 + kb0 + j;
                afrag[8 + j] = (k < K) ? (__bf16)Arow[k] : (__bf16)0.f;
            }
        }

        // software-pipelined B loads: issue load(t+1) before wmma(t)
        v16bf bcur = *reinterpret_cast<const v16bf*>(&Bs[buf][bn][bkb]);
#pragma unroll
        for (int t = 0; t < NT; ++t) {
            v16bf bnext;
            if (t + 1 < NT)
                bnext = *reinterpret_cast<const v16bf*>(
                    &Bs[buf][(t + 1) * 16 + bn][bkb]);
            acc[t] = __builtin_amdgcn_wmma_f32_16x16x32_bf16(
                false, afrag, false, bcur, (short)0, acc[t], false, false);
            bcur = bnext;
        }
        __syncthreads();
    }

    // epilogue
    const int rowHalf = (lane >> 4) * 8;
#pragma unroll
    for (int t = 0; t < NT; ++t) {
        const int col = nBlock + t * 16 + bn;
        if (col >= N) continue;
        const float bv = bias ? bias[col] : 0.f;
#pragma unroll
        for (int rr = 0; rr < 8; ++rr) {
            const int row = mBase + rowHalf + rr;
            if (row >= M) continue;
            float v = acc[t][rr] + bv;
            if (act == ACT_SOFTPLUS) v = (v > 20.f) ? v : log1pf(__expf(v));
            if (resid) v += resid[(size_t)row * ldr + col];
            C[(size_t)row * ldc + col] = v;
        }
    }
}

// ---------------------------------------------------------------------------
// Patch embed: 4x4/stride-4 conv, NCHW (2,3,256,256) -> (b, 64*64, 96)
// ---------------------------------------------------------------------------
__global__ void patch_embed_kernel(const float* __restrict__ x,
                                   const float* __restrict__ w,
                                   const float* __restrict__ b,
                                   float* __restrict__ out,
                                   int Bn, int H, int Cout)
{
    const size_t total = (size_t)Bn * H * H * Cout;
    const size_t i = (size_t)blockIdx.x * blockDim.x + threadIdx.x;
    if (i >= total) return;
    const int oc = (int)(i % Cout);
    size_t t = i / Cout;
    const int ow = (int)(t % H); t /= H;
    const int oh = (int)(t % H);
    const int bb = (int)(t / H);

    float acc = b[oc];
#pragma unroll
    for (int ic = 0; ic < 3; ++ic)
#pragma unroll
        for (int kh = 0; kh < 4; ++kh)
#pragma unroll
            for (int kw = 0; kw < 4; ++kw)
                acc += w[((oc * 3 + ic) * 4 + kh) * 4 + kw] *
                       x[(((size_t)bb * 3 + ic) * 256 + (oh * 4 + kh)) * 256 +
                         (ow * 4 + kw)];
    out[i] = acc;
}

// ---------------------------------------------------------------------------
// LayerNorm over last dim D; one 128-thread block per row
// ---------------------------------------------------------------------------
__global__ __launch_bounds__(128)
void layernorm_kernel(const float* __restrict__ in,
                      const float* __restrict__ g,
                      const float* __restrict__ b,
                      float* __restrict__ out, int D)
{
    const float* __restrict__ row = in  + (size_t)blockIdx.x * D;
    float*       __restrict__ oro = out + (size_t)blockIdx.x * D;
    const int tid = threadIdx.x;
    float s = 0.f, s2 = 0.f;
    for (int j = tid; j < D; j += 128) {
        const float v = row[j];
        s += v; s2 += v * v;
    }
    __shared__ float sh0[128], sh1[128];
    sh0[tid] = s; sh1[tid] = s2;
    __syncthreads();
    for (int off = 64; off > 0; off >>= 1) {
        if (tid < off) { sh0[tid] += sh0[tid + off]; sh1[tid] += sh1[tid + off]; }
        __syncthreads();
    }
    const float mean = sh0[0] / D;
    const float var  = sh1[0] / D - mean * mean;
    const float inv  = rsqrtf(var + 1e-5f);
    for (int j = tid; j < D; j += 128)
        oro[j] = (row[j] - mean) * inv * g[j] + b[j];
}

// ---------------------------------------------------------------------------
// Fused 4-direction combine + LayerNorm.  One block per output row (b,l).
//   y(b,l=(r,c)) = y0[l] + y1[c*hw+hw-1-r] + y2[L-1-l] + y3[L-1-(c*hw+hw-1-r)]
// ---------------------------------------------------------------------------
__global__ __launch_bounds__(128)
void combine_ln_kernel(const float* __restrict__ ys,
                       const float* __restrict__ g,
                       const float* __restrict__ bb,
                       float* __restrict__ out,
                       int B, int L, int hw, int d)
{
    const int row = blockIdx.x;           // over B*L
    const int b = row / L, l = row % L;
    const int rr = l / hw, cc = l % hw;
    const int l2 = cc * hw + (hw - 1 - rr);
    const int l3 = L - 1 - l;
    const int l4 = L - 1 - l2;
    const size_t sd = (size_t)L * d;
    const float* __restrict__ r0 = ys + (size_t)b           * sd + (size_t)l  * d;
    const float* __restrict__ r1 = ys + (size_t)(B + b)     * sd + (size_t)l2 * d;
    const float* __restrict__ r2 = ys + (size_t)(2 * B + b) * sd + (size_t)l3 * d;
    const float* __restrict__ r3 = ys + (size_t)(3 * B + b) * sd + (size_t)l4 * d;

    __shared__ float buf[768];            // d <= 768
    __shared__ float sh0[128], sh1[128];
    const int tid = threadIdx.x;
    float s = 0.f, s2 = 0.f;
    for (int j = tid; j < d; j += 128) {
        const float v = r0[j] + r1[j] + r2[j] + r3[j];
        buf[j] = v; s += v; s2 += v * v;
    }
    sh0[tid] = s; sh1[tid] = s2;
    __syncthreads();
    for (int off = 64; off > 0; off >>= 1) {
        if (tid < off) { sh0[tid] += sh0[tid + off]; sh1[tid] += sh1[tid + off]; }
        __syncthreads();
    }
    const float mean = sh0[0] / d;
    const float var  = sh1[0] / d - mean * mean;
    const float inv  = rsqrtf(var + 1e-5f);
    float* __restrict__ orow = out + (size_t)row * d;
    for (int j = tid; j < d; j += 128)
        orow[j] = (buf[j] - mean) * inv * g[j] + bb[j];
}

// ---------------------------------------------------------------------------
// Build xs (4B, L, d) from x (B, L, d): [x1f, x2f, rev(x1f), rev(x2f)]
// ---------------------------------------------------------------------------
__global__ void gather_dirs_kernel(const float* __restrict__ x,
                                   float* __restrict__ xs,
                                   int B, int L, int hw, int d)
{
    const size_t total = (size_t)4 * B * L * d;
    const size_t i = (size_t)blockIdx.x * blockDim.x + threadIdx.x;
    if (i >= total) return;
    const int c = (int)(i % d);
    size_t t = i / d;
    const int l = (int)(t % L);
    const int s = (int)(t / L);
    const int dir = s / B, b = s % B;
    int lsrc;
    if (dir == 0) lsrc = l;
    else if (dir == 1) { const int r = l / hw, cc = l % hw; lsrc = (hw - 1 - cc) * hw + r; }
    else if (dir == 2) lsrc = L - 1 - l;
    else { const int l2 = L - 1 - l; const int r = l2 / hw, cc = l2 % hw; lsrc = (hw - 1 - cc) * hw + r; }
    xs[i] = x[((size_t)b * L + lsrc) * d + c];
}

// ---------------------------------------------------------------------------
// Depthwise causal conv (k=3) + bias + SiLU.  Input = xz[..., :din] with row
// stride 2*din; output xi (S, L, din) contiguous.
// ---------------------------------------------------------------------------
__global__ void dwconv_silu_kernel(const float* __restrict__ xz,
                                   const float* __restrict__ w,
                                   const float* __restrict__ b,
                                   float* __restrict__ xi,
                                   int S, int L, int din)
{
    const size_t total = (size_t)S * L * din;
    const size_t i = (size_t)blockIdx.x * blockDim.x + threadIdx.x;
    if (i >= total) return;
    const int c = (int)(i % din);
    size_t t = i / din;
    const int l = (int)(t % L);
    const int s = (int)(t / L);
    const size_t st = (size_t)2 * din;
    const float* __restrict__ base = xz + ((size_t)s * L) * st + c;
    float acc = b[c];
#pragma unroll
    for (int j = 0; j < 3; ++j) {
        const int ll = l - 2 + j;
        if (ll >= 0) acc += w[c * 3 + j] * base[(size_t)ll * st];
    }
    xi[i] = silu_f(acc);
}

// ---------------------------------------------------------------------------
// Chunked selective scan.  CHUNK = 64 tokens.
//   Pass 1: per chunk, a[n] = prod(dA), hE[n] = chunk scan from h=0.
//   Pass 2: prefix-combine chunk states -> h0 (initial h of each chunk).
//   Pass 3: replay each chunk from h0, emit y = (h.C + D*x) * silu(z).
// ---------------------------------------------------------------------------
__global__ __launch_bounds__(256)
void scan_pass1_kernel(const float* __restrict__ xi,
                       const float* __restrict__ dt,
                       const float* __restrict__ xdbl,
                       const float* __restrict__ A_log,
                       float* __restrict__ aP, float* __restrict__ hE,
                       int L, int din, int xst, int r, int nch)
{
    const int s  = blockIdx.x;
    const int c  = blockIdx.y;
    const int ch = blockIdx.z * 256 + threadIdx.x;
    const int l0 = c * 64;
    __shared__ float BC[2][16];

    float Avec[16], a[16], h[16];
#pragma unroll
    for (int n = 0; n < 16; ++n) { Avec[n] = 0.f; a[n] = 1.f; h[n] = 0.f; }
    if (ch < din)
#pragma unroll
        for (int n = 0; n < 16; ++n) Avec[n] = -__expf(A_log[ch * 16 + n]);

    if (threadIdx.x < 16)
        BC[0][threadIdx.x] = xdbl[((size_t)s * L + l0) * xst + r + threadIdx.x];
    __syncthreads();

    for (int t = 0; t < 64; ++t) {
        const int pb = t & 1;
        if ((t + 1 < 64) && threadIdx.x < 16)
            BC[pb ^ 1][threadIdx.x] =
                xdbl[((size_t)s * L + l0 + t + 1) * xst + r + threadIdx.x];
        if (ch < din) {
            const size_t tok = (size_t)s * L + l0 + t;
            const float dtv = dt[tok * din + ch];
            const float xv  = xi[tok * din + ch];
#pragma unroll
            for (int n = 0; n < 16; ++n) {
                const float dA = __expf(dtv * Avec[n]);
                a[n] *= dA;
                h[n] = dA * h[n] + dtv * BC[pb][n] * xv;
            }
        }
        __syncthreads();
    }
    if (ch < din) {
        const size_t base = (((size_t)s * nch + c) * din + ch) * 16;
#pragma unroll
        for (int n = 0; n < 16; ++n) { aP[base + n] = a[n]; hE[base + n] = h[n]; }
    }
}

__global__ void scan_pass2_kernel(const float* __restrict__ aP,
                                  const float* __restrict__ hE,
                                  float* __restrict__ h0,
                                  int S, int din, int nch)
{
    const int i = blockIdx.x * blockDim.x + threadIdx.x;
    if (i >= S * din * 16) return;
    const int n = i & 15;
    const int t = i >> 4;
    const int ch = t % din;
    const int s  = t / din;
    float h = 0.f;
    for (int c = 0; c < nch; ++c) {
        const size_t idx = (((size_t)s * nch + c) * din + ch) * 16 + n;
        h0[idx] = h;
        h = aP[idx] * h + hE[idx];
    }
}

__global__ __launch_bounds__(256)
void scan_pass3_kernel(const float* __restrict__ xi,
                       const float* __restrict__ dt,
                       const float* __restrict__ xdbl,
                       const float* __restrict__ xz,
                       const float* __restrict__ A_log,
                       const float* __restrict__ Dp,
                       const float* __restrict__ h0,
                       float* __restrict__ y,
                       int L, int din, int xst, int r, int d4, int nch)
{
    const int s  = blockIdx.x;
    const int c  = blockIdx.y;
    const int ch = blockIdx.z * 256 + threadIdx.x;
    const int l0 = c * 64;
    __shared__ float BC[2][32];

    float Avec[16], h[16];
    float Dch = 0.f;
#pragma unroll
    for (int n = 0; n < 16; ++n) { Avec[n] = 0.f; h[n] = 0.f; }
    if (ch < din) {
        Dch = Dp[ch];
        const size_t base = (((size_t)s * nch + c) * din + ch) * 16;
#pragma unroll
        for (int n = 0; n < 16; ++n) {
            Avec[n] = -__expf(A_log[ch * 16 + n]);
            h[n] = h0[base + n];
        }
    }

    if (threadIdx.x < 32)
        BC[0][threadIdx.x] = xdbl[((size_t)s * L + l0) * xst + r + threadIdx.x];
    __syncthreads();

    for (int t = 0; t < 64; ++t) {
        const int pb = t & 1;
        if ((t + 1 < 64) && threadIdx.x < 32)
            BC[pb ^ 1][threadIdx.x] =
                xdbl[((size_t)s * L + l0 + t + 1) * xst + r + threadIdx.x];
        if (ch < din) {
            const size_t tok = (size_t)s * L + l0 + t;
            const float dtv = dt[tok * din + ch];
            const float xv  = xi[tok * din + ch];
            float acc = 0.f;
#pragma unroll
            for (int n = 0; n < 16; ++n) {
                const float dA = __expf(dtv * Avec[n]);
                h[n] = dA * h[n] + dtv * BC[pb][n] * xv;
                acc += h[n] * BC[pb][16 + n];
            }
            const float zv = xz[tok * (size_t)d4 + din + ch];
            y[tok * din + ch] = (acc + Dch * xv) * silu_f(zv);
        }
        __syncthreads();
    }
}

// ---------------------------------------------------------------------------
// Patch merging gather: (B,hw,hw,d) -> (B, hw/2*hw/2, 4d)
// ---------------------------------------------------------------------------
__global__ void pm_gather_kernel(const float* __restrict__ x,
                                 float* __restrict__ out,
                                 int B, int hw, int d)
{
    const int h2 = hw >> 1;
    const size_t total = (size_t)B * h2 * h2 * 4 * d;
    const size_t i = (size_t)blockIdx.x * blockDim.x + threadIdx.x;
    if (i >= total) return;
    const int c4 = (int)(i % (4 * d));
    size_t t = i / (4 * d);
    const int cc = (int)(t % h2); t /= h2;
    const int rr = (int)(t % h2);
    const int bb = (int)(t / h2);
    const int q = c4 / d, ch = c4 % d;
    const int sr = 2 * rr + (q & 1);
    const int sc = 2 * cc + (q >> 1);
    out[i] = x[(((size_t)bb * hw + sr) * hw + sc) * d + ch];
}

// ---------------------------------------------------------------------------
// Mean over T tokens: (B,T,D) -> (B,D)
// ---------------------------------------------------------------------------
__global__ void mean_tokens_kernel(const float* __restrict__ in,
                                   float* __restrict__ out,
                                   int B, int T, int D)
{
    const int i = blockIdx.x * blockDim.x + threadIdx.x;
    if (i >= B * D) return;
    const int b = i / D, c = i % D;
    float s = 0.f;
    for (int t = 0; t < T; ++t) s += in[((size_t)b * T + t) * D + c];
    out[i] = s / (float)T;
}

// ---------------------------------------------------------------------------
// Tiny classifier matvec
// ---------------------------------------------------------------------------
__global__ void head_kernel(const float* __restrict__ pooled,
                            const float* __restrict__ w,
                            const float* __restrict__ bias,
                            float* __restrict__ out,
                            int B, int D, int NC)
{
    const int i = threadIdx.x;
    if (i >= B * NC) return;
    const int b = i / NC, o = i % NC;
    float s = bias[o];
    for (int j = 0; j < D; ++j) s += pooled[(size_t)b * D + j] * w[(size_t)o * D + j];
    out[i] = s;
}

// ---------------------------------------------------------------------------
// Host orchestration
// ---------------------------------------------------------------------------
static const int DIMS_[4]   = {96, 192, 384, 768};
static const int DEPTHS_[4] = {2, 2, 5, 2};

struct MP { const float *ipw,*cw,*cb,*xpw,*dpw,*dpb,*alog,*dd,*opw,*opb; };
struct BP { MP m; const float *g,*b,*pw,*pb,*bg,*bb; };

static inline void launch_gemm(hipStream_t st, const float* A, int lda,
                               const float* W, const float* bias,
                               const float* resid, int ldr,
                               float* C, int ldc, int M, int N, int K, int act)
{
    if (N > 64) {
        dim3 g((unsigned)((N + 127) / 128), (unsigned)((M + 127) / 128));
        gemm_wmma_kernel<8><<<g, dim3(256), 0, st>>>(A, lda, W, bias, resid, ldr,
                                                     C, ldc, M, N, K, act);
    } else {
        dim3 g((unsigned)((N + 63) / 64), (unsigned)((M + 127) / 128));
        gemm_wmma_kernel<4><<<g, dim3(256), 0, st>>>(A, lda, W, bias, resid, ldr,
                                                     C, ldc, M, N, K, act);
    }
}

extern "C" void kernel_launch(void* const* d_in, const int* in_sizes, int n_in,
                              void* d_out, int out_size, void* d_ws, size_t ws_size,
                              hipStream_t stream)
{
    (void)in_sizes; (void)n_in; (void)out_size; (void)ws_size;
    // ---- parse inputs (setup_inputs insertion order) ----
    int p = 0;
    auto nx = [&]() -> const float* { return (const float*)d_in[p++]; };
    const float* Xin  = nx();
    const float* pe_w = nx(); const float* pe_b = nx();
    const float* pe_g = nx(); const float* pe_be = nx();
    BP blk[11]; int nb = 0;
    for (int s = 0; s < 4; ++s)
        for (int j = 0; j < DEPTHS_[s]; ++j) {
            BP bp;
            bp.m.ipw = nx(); bp.m.cw  = nx(); bp.m.cb  = nx();
            bp.m.xpw = nx(); bp.m.dpw = nx(); bp.m.dpb = nx();
            bp.m.alog= nx(); bp.m.dd  = nx();
            bp.m.opw = nx(); bp.m.opb = nx();
            bp.g = nx(); bp.b = nx(); bp.pw = nx(); bp.pb = nx();
            bp.bg = nx(); bp.bb = nx();
            blk[nb++] = bp;
        }
    const float *dn_g[3], *dn_b[3], *dn_w[3];
    for (int i = 0; i < 3; ++i) { dn_g[i] = nx(); dn_b[i] = nx(); dn_w[i] = nx(); }
    const float* cls_g = nx(); const float* cls_b = nx();
    const float* cls_w = nx(); const float* cls_bias = nx();

    // ---- workspace carve-out (f32 elements) ----
    float* ws = (float*)d_ws;
    size_t off = 0;
    auto alloc = [&](size_t n) { float* q = ws + off; off += n; return q; };
    float* xb0 = alloc(786432);     // (B,L,d) ping
    float* xb1 = alloc(786432);     // (B,L,d) pong
    float* xsB = alloc(3145728);    // (4B,L,d)
    float* xzB = alloc(12582912);   // (4B,L,4d)
    float* xiB = alloc(6291456);    // (4B,L,2d)
    float* xdB = alloc(1310720);    // (4B,L,r+32)
    float* dtB = alloc(6291456);    // (4B,L,2d)
    float* yB  = alloc(6291456);    // (4B,L,2d)
    float* moB = alloc(3145728);    // (4B,L,d)
    float* lnB = alloc(786432);     // LN output temp
    float* ctB = alloc(786432);     // patch-merge cat / patch-embed temp
    float* aPB = alloc(1572864);    // scan chunk products
    float* hEB = alloc(1572864);    // scan chunk end-states
    float* h0B = alloc(1572864);    // scan chunk init-states
    float* pooled = alloc(2048);

    float* outF = (float*)d_out;
    const size_t outOffX  = 12;
    const size_t outOffF0 = outOffX  + 98304;
    const size_t outOffF1 = outOffF0 + 786432;
    const size_t outOffF2 = outOffF1 + 393216;
    const size_t featOff[3]  = {outOffF0, outOffF1, outOffF2};

    auto ew = [&](size_t total) { return dim3((unsigned)((total + 255) / 256)); };

    // ---- patch embed ----
    {
        const size_t tot = (size_t)2 * 64 * 64 * 96;
        patch_embed_kernel<<<ew(tot), 256, 0, stream>>>(Xin, pe_w, pe_b, ctB, 2, 64, 96);
        layernorm_kernel<<<2 * 4096, 128, 0, stream>>>(ctB, pe_g, pe_be, xb0, 96);
    }

    float* xcur = xb0;
    float* xalt = xb1;
    int d = 96, hw = 64;
    int bi = 0;

    for (int st = 0; st < 4; ++st) {
        const int L   = hw * hw;
        const int din = 2 * d;
        const int r   = d / 16;
        const int xst = r + 32;
        const int S   = 8;            // 4 dirs * batch 2
        const int ML  = S * L;
        const int Mb  = 2 * L;        // batch rows
        const int nch = L / 64;

        for (int j = 0; j < DEPTHS_[st]; ++j) {
            const BP& bp = blk[bi++];
            gather_dirs_kernel<<<ew((size_t)ML * d), 256, 0, stream>>>(
                xcur, xsB, 2, L, hw, d);
            // in_proj
            launch_gemm(stream, xsB, d, bp.m.ipw, nullptr, nullptr, 0,
                        xzB, 4 * d, ML, 4 * d, d, ACT_NONE);
            // depthwise causal conv + SiLU
            dwconv_silu_kernel<<<ew((size_t)ML * din), 256, 0, stream>>>(
                xzB, bp.m.cw, bp.m.cb, xiB, S, L, din);
            // x_proj
            launch_gemm(stream, xiB, din, bp.m.xpw, nullptr, nullptr, 0,
                        xdB, xst, ML, xst, din, ACT_NONE);
            // dt_proj + softplus
            launch_gemm(stream, xdB, xst, bp.m.dpw, bp.m.dpb, nullptr, 0,
                        dtB, din, ML, din, r, ACT_SOFTPLUS);
            // chunked selective scan
            {
                dim3 g1((unsigned)S, (unsigned)nch, (unsigned)((din + 255) / 256));
                scan_pass1_kernel<<<g1, 256, 0, stream>>>(
                    xiB, dtB, xdB, bp.m.alog, aPB, hEB, L, din, xst, r, nch);
                scan_pass2_kernel<<<ew((size_t)S * din * 16), 256, 0, stream>>>(
                    aPB, hEB, h0B, S, din, nch);
                scan_pass3_kernel<<<g1, 256, 0, stream>>>(
                    xiB, dtB, xdB, xzB, bp.m.alog, bp.m.dd, h0B, yB,
                    L, din, xst, r, 4 * d, nch);
            }
            // out_proj
            launch_gemm(stream, yB, din, bp.m.opw, bp.m.opb, nullptr, 0,
                        moB, d, ML, d, din, ACT_NONE);
            // fused combine + LN
            combine_ln_kernel<<<Mb, 128, 0, stream>>>(moB, bp.g, bp.b, lnB,
                                                      2, L, hw, d);
            // proj + residual
            launch_gemm(stream, lnB, d, bp.pw, bp.pb, xcur, d,
                        xalt, d, Mb, d, d, ACT_NONE);
            { float* t = xcur; xcur = xalt; xalt = t; }
            if (j < DEPTHS_[st] - 1) {
                layernorm_kernel<<<Mb, 128, 0, stream>>>(xcur, bp.bg, bp.bb, xalt, d);
                { float* t = xcur; xcur = xalt; xalt = t; }
            }
        }

        // emit features / final x
        const size_t feBytes = (size_t)2 * L * d * sizeof(float);
        if (st < 3)
            hipMemcpyAsync(outF + featOff[st], xcur, feBytes,
                           hipMemcpyDeviceToDevice, stream);
        else
            hipMemcpyAsync(outF + outOffX, xcur, feBytes,
                           hipMemcpyDeviceToDevice, stream);

        // patch merging
        if (st < 3) {
            const int h2 = hw / 2;
            pm_gather_kernel<<<ew((size_t)2 * h2 * h2 * 4 * d), 256, 0, stream>>>(
                xcur, ctB, 2, hw, d);
            layernorm_kernel<<<2 * h2 * h2, 128, 0, stream>>>(
                ctB, dn_g[st], dn_b[st], lnB, 4 * d);
            launch_gemm(stream, lnB, 4 * d, dn_w[st], nullptr, nullptr, 0,
                        xalt, 2 * d, 2 * h2 * h2, 2 * d, 4 * d, ACT_NONE);
            { float* t = xcur; xcur = xalt; xalt = t; }
            d *= 2; hw /= 2;
        }
    }

    // ---- classifier head ----
    layernorm_kernel<<<2 * 64, 128, 0, stream>>>(xcur, cls_g, cls_b, lnB, 768);
    mean_tokens_kernel<<<ew(2 * 768), 256, 0, stream>>>(lnB, pooled, 2, 64, 768);
    head_kernel<<<1, 32, 0, stream>>>(pooled, cls_w, cls_bias, outF, 2, 768, 6);
}
